// AttentionLayer_60894046322746
// MI455X (gfx1250) — compile-verified, matrix operand order checked by reading, not run on hardware
//
#include <hip/hip_runtime.h>

// AttentionLayer for MI455X (gfx1250): bf16 WMMA pipeline.
// cvt (+weight transpose) -> qkv GEMM (32x64 wave tiles) + rotary epilogue
//   -> flash attention (async LDS staging, online softmax)
//   -> proj GEMM (32x64 wave tiles).

namespace {

constexpr int Bsz  = 2;
constexpr int Sq   = 2048;
constexpr int HID  = 1024;
constexpr int DH   = 64;
constexpr int NH   = 16;
constexpr int ROT  = 32;
constexpr int NQKV = 3 * NH * DH;   // 3072
constexpr int MROWS = Bsz * Sq;     // 4096

typedef __attribute__((ext_vector_type(16))) __bf16 v16bf;
typedef __attribute__((ext_vector_type(8)))  float  v8f;

union FragBF {
  v16bf v;
  unsigned short u[16];
  unsigned int   d[8];
};

__device__ __forceinline__ unsigned short f2bf(float f) {
  unsigned int u = __float_as_uint(f);
  u += 0x7FFFu + ((u >> 16) & 1u);   // round-to-nearest-even
  return (unsigned short)(u >> 16);
}

__device__ __forceinline__ v8f wmma_bf16(v16bf a, v16bf b, v8f c) {
  // D = A(16x32 bf16) * B(32x16 bf16) + C(16x16 f32)
  return __builtin_amdgcn_wmma_f32_16x16x32_bf16(
      false, a, false, b, (short)0, c, false, false);
}

// K index inside a 16x32 bf16 A (or 32x16 B) fragment for VGPR slot v,
// per ISA 7.12.2: lanes 0-15 hold K={2v,2v+1} (+8 for v>=4); lanes 16-31 +8.
__device__ __forceinline__ int frag_k(int v, int hlf) {
  return ((v >> 2) << 4) + (hlf << 3) + ((v & 3) << 1);
}

// Async global->LDS 16B copy per lane (tracked with ASYNCcnt).
__device__ __forceinline__ void async_b128(const unsigned short* gsrc,
                                           unsigned short* lds_dst) {
  unsigned int ldsa = (unsigned int)(size_t)lds_dst;        // low 32 = LDS addr
  unsigned long long ga = (unsigned long long)(size_t)gsrc;
  asm volatile("global_load_async_to_lds_b128 %0, %1, off"
               :: "v"(ldsa), "v"(ga) : "memory");
}

// ---------------------------------------------------------------------------
// f32 -> bf16 conversion; weights are also transposed to N x K (K-contiguous)
// so every WMMA B-fragment load is two global_load_b128.
// ---------------------------------------------------------------------------
__global__ void cvt_kernel(const float* __restrict__ x,
                           const float* __restrict__ wq,
                           const float* __restrict__ wp,
                           unsigned short* __restrict__ xb,    // 4096 x 1024
                           unsigned short* __restrict__ wqtb,  // 3072 x 1024
                           unsigned short* __restrict__ wptb)  // 1024 x 1024
{
  size_t tid = (size_t)blockIdx.x * blockDim.x + threadIdx.x;
  size_t stride = (size_t)gridDim.x * blockDim.x;
  for (size_t i = tid; i < (size_t)MROWS * HID; i += stride) xb[i] = f2bf(x[i]);
  for (size_t i = tid; i < (size_t)NQKV * HID; i += stride) {
    size_t n = i >> 10, k = i & (HID - 1);
    wqtb[i] = f2bf(wq[k * NQKV + n]);
  }
  for (size_t i = tid; i < (size_t)HID * HID; i += stride) {
    size_t n = i >> 10, k = i & (HID - 1);
    wptb[i] = f2bf(wp[k * HID + n]);
  }
}

// ---------------------------------------------------------------------------
// QKV GEMM: (4096 x 1024) x (1024 x 3072) + bias, rotary, scale, layout split.
// Block = 4 waves, each wave computes a 32x64 tile (2 A-frags x 4 B-frags,
// 8 WMMA per K-step; each B fragment reused twice).
// ---------------------------------------------------------------------------
__global__ __launch_bounds__(128) void qkv_gemm_kernel(
    const unsigned short* __restrict__ xb,    // 4096 x 1024  bf16
    const unsigned short* __restrict__ wqtb,  // 3072 x 1024  bf16 (W^T)
    const float* __restrict__ qkv_bias,       // 3072         f32
    const float* __restrict__ sinu,           // 2 x 2048 x 32 f32
    unsigned short* __restrict__ Qb,          // [b][h][s][d] bf16 (scaled)
    unsigned short* __restrict__ Kb,          // [b][h][s][d] bf16
    unsigned short* __restrict__ VTb) {       // [b][h][d][s] bf16 (transposed)
  const int lane = threadIdx.x & 31;
  const int wave = threadIdx.x >> 5;
  const int nloc = lane & 15;
  const int hlf  = lane >> 4;
  const int m0   = blockIdx.y * 128 + wave * 32;
  const int c0   = blockIdx.x * 64;

  v8f acc[2][4] = {{{}, {}, {}, {}}, {{}, {}, {}, {}}};

  for (int kb = 0; kb < HID; kb += 32) {
    FragBF a[2];
#pragma unroll
    for (int r = 0; r < 2; ++r)
#pragma unroll
      for (int v = 0; v < 8; ++v) {
        int kk = frag_k(v, hlf);
        a[r].d[v] = *(const unsigned int*)(
            xb + (size_t)(m0 + r * 16 + nloc) * HID + kb + kk);
      }
#pragma unroll
    for (int t = 0; t < 4; ++t) {
      FragBF bw;  // B[k][n] = WT[n][k], contiguous in k
#pragma unroll
      for (int v = 0; v < 8; ++v) {
        int kk = frag_k(v, hlf);
        bw.d[v] = *(const unsigned int*)(
            wqtb + (size_t)(c0 + t * 16 + nloc) * HID + kb + kk);
      }
      acc[0][t] = wmma_bf16(a[0].v, bw.v, acc[0][t]);
      acc[1][t] = wmma_bf16(a[1].v, bw.v, acc[1][t]);
    }
  }

  // Epilogue: bias + rotary (elementwise factor) + scale; scatter to Q/K/V^T.
  const int b     = m0 / Sq;
  const int sbase = m0 % Sq;
#pragma unroll
  for (int t = 0; t < 4; ++t) {
    int c   = c0 + t * 16 + nloc;   // global qkv column [0, 3072)
    int idx = c >> 6;               // 0..47 : q heads 0-15, k 16-31, v 32-47
    int d   = c & 63;
    float bv = qkv_bias[c];
#pragma unroll
    for (int r = 0; r < 2; ++r)
#pragma unroll
      for (int v = 0; v < 8; ++v) {
        int srow  = sbase + r * 16 + v + 8 * hlf;
        float val = acc[r][t][v] + bv;
        if (idx < 2 * NH && d < ROT) {
          float sn = sinu[srow * ROT + d];
          float cs = sinu[Sq * ROT + srow * ROT + d];
          val *= (d & 1) ? (cs + sn) : (cs - sn);
        }
        if (idx < NH) {
          val *= 0.125f;  // 1/sqrt(64) folded into Q
          Qb[((size_t)(b * NH + idx) * Sq + srow) * DH + d] = f2bf(val);
        } else if (idx < 2 * NH) {
          Kb[((size_t)(b * NH + (idx - NH)) * Sq + srow) * DH + d] = f2bf(val);
        } else {
          VTb[((size_t)(b * NH + (idx - 2 * NH)) * DH + d) * Sq + srow] =
              f2bf(val);
        }
      }
  }
}

// ---------------------------------------------------------------------------
// Flash attention: block = 4 waves; each wave owns 16 query rows.
// Per 64-key chunk: async-stage K/V^T tiles to LDS (shared by all 4 waves),
// 8 WMMA (QK^T) + bias + online softmax + 8 WMMA (PV).
// ---------------------------------------------------------------------------
__global__ __launch_bounds__(128) void attn_kernel(
    const unsigned short* __restrict__ Qb,    // [b][h][s][d]
    const unsigned short* __restrict__ Kb,    // [b][h][s][d]
    const unsigned short* __restrict__ VTb,   // [b][h][d][s]
    const float* __restrict__ bias,           // [b][1][s][s]
    unsigned short* __restrict__ ctxb) {      // [b*s][h*64+d] bf16
  __shared__ __align__(16) unsigned short kt[64 * 64];        // [key][d]  8 KB
  __shared__ __align__(16) unsigned short vt[64 * 64];        // [d][key]  8 KB
  __shared__ __align__(16) unsigned short plds[4 * 16 * 64];  // P tiles   8 KB

  const int tid  = threadIdx.x;
  const int lane = tid & 31;
  const int wave = tid >> 5;
  const int nloc = lane & 15;
  const int hlf  = lane >> 4;
  const int h    = blockIdx.y;
  const int b    = blockIdx.z;
  const int q0   = blockIdx.x * 64 + wave * 16;
  const int wbase = wave * 16 * 64;

  const unsigned short* Qh  = Qb  + (size_t)(b * NH + h) * Sq * DH;
  const unsigned short* Kh  = Kb  + (size_t)(b * NH + h) * Sq * DH;
  const unsigned short* VTh = VTb + (size_t)(b * NH + h) * DH * Sq;
  const float* biasb = bias + (size_t)b * Sq * Sq;

  // Q fragments (d 0-31, 32-63), contiguous pairs in d.
  FragBF qa0, qa1;
#pragma unroll
  for (int v = 0; v < 8; ++v) {
    int kk = frag_k(v, hlf);
    const unsigned short* p = Qh + (size_t)(q0 + nloc) * DH + kk;
    qa0.d[v] = *(const unsigned int*)p;
    qa1.d[v] = *(const unsigned int*)(p + 32);
  }

  v8f acc[4] = {{}, {}, {}, {}};
  float rm[8], rl[8];
#pragma unroll
  for (int v = 0; v < 8; ++v) { rm[v] = -1e30f; rl[v] = 0.0f; }

  for (int kb = 0; kb < Sq; kb += 64) {
    // ---- async-stage K chunk (8 KB contiguous) and V^T chunk (64 x 128B) ----
    {
      const unsigned short* ksrc = Kh + (size_t)kb * DH;
#pragma unroll
      for (int j = 0; j < 4; ++j) {
        int el = (tid + j * 128) * 8;          // 16B per lane
        async_b128(ksrc + el, &kt[el]);
      }
      const int drow = tid >> 1;               // 2 threads per V^T row
      const int seg  = (tid & 1) * 32;         // 64B half-row
      const unsigned short* vsrc = VTh + (size_t)drow * Sq + kb + seg;
#pragma unroll
      for (int j = 0; j < 4; ++j)
        async_b128(vsrc + j * 8, &vt[drow * 64 + seg + j * 8]);
    }
    asm volatile("s_wait_asynccnt 0" ::: "memory");
    __syncthreads();

    // ---- scores: 4 tiles of 16x16 over keys kb..kb+63 ----
    v8f sc[4];
#pragma unroll
    for (int t = 0; t < 4; ++t) {
      v8f c = {};
      FragBF bk0, bk1;  // B[kd][kn] = K[key][d], contiguous in d (from LDS)
#pragma unroll
      for (int v = 0; v < 8; ++v) {
        int kk = frag_k(v, hlf);
        const unsigned short* p = &kt[(t * 16 + nloc) * 64 + kk];
        bk0.d[v] = *(const unsigned int*)p;
        bk1.d[v] = *(const unsigned int*)(p + 32);
      }
      c = wmma_bf16(qa0.v, bk0.v, c);
      c = wmma_bf16(qa1.v, bk1.v, c);
#pragma unroll
      for (int v = 0; v < 8; ++v) {
        int m = v + 8 * hlf;
        c[v] += biasb[(size_t)(q0 + m) * Sq + kb + t * 16 + nloc];
      }
      sc[t] = c;
    }

    // ---- online softmax (rows spread over 16-lane halves) ----
    v8f alphav;
#pragma unroll
    for (int v = 0; v < 8; ++v) {
      float tm = fmaxf(fmaxf(sc[0][v], sc[1][v]), fmaxf(sc[2][v], sc[3][v]));
      tm = fmaxf(tm, __shfl_xor(tm, 1, 32));
      tm = fmaxf(tm, __shfl_xor(tm, 2, 32));
      tm = fmaxf(tm, __shfl_xor(tm, 4, 32));
      tm = fmaxf(tm, __shfl_xor(tm, 8, 32));
      float mnew  = fmaxf(rm[v], tm);
      float alpha = __expf(rm[v] - mnew);
      float rsum = 0.0f;
#pragma unroll
      for (int t = 0; t < 4; ++t) {
        float p = __expf(sc[t][v] - mnew);
        sc[t][v] = p;
        rsum += p;
      }
      rsum += __shfl_xor(rsum, 1, 32);
      rsum += __shfl_xor(rsum, 2, 32);
      rsum += __shfl_xor(rsum, 4, 32);
      rsum += __shfl_xor(rsum, 8, 32);
      rl[v] = rl[v] * alpha + rsum;
      rm[v] = mnew;
      alphav[v] = alpha;
    }
#pragma unroll
    for (int dt = 0; dt < 4; ++dt) acc[dt] *= alphav;

    // ---- stage P (C-layout) into LDS, reload in A-layout (wave-private) ----
#pragma unroll
    for (int t = 0; t < 4; ++t)
#pragma unroll
      for (int v = 0; v < 8; ++v) {
        int m = v + 8 * hlf;
        plds[wbase + m * 64 + t * 16 + nloc] = f2bf(sc[t][v]);
      }
    asm volatile("s_wait_dscnt 0" ::: "memory");  // wave-local store->load

    FragBF p0, p1;
#pragma unroll
    for (int v = 0; v < 8; ++v) {
      int kk  = frag_k(v, hlf);
      int off = wbase + nloc * 64 + kk;  // A-frag row = lane & 15
      p0.u[2 * v]     = plds[off];
      p0.u[2 * v + 1] = plds[off + 1];
      p1.u[2 * v]     = plds[off + 32];
      p1.u[2 * v + 1] = plds[off + 33];
    }

    // ---- ctx += P * V  (V^T tile in LDS -> contiguous in key) ----
#pragma unroll
    for (int dt = 0; dt < 4; ++dt) {
      FragBF bv0, bv1;  // B[key][d] = vt[d][key]
#pragma unroll
      for (int v = 0; v < 8; ++v) {
        int kk = frag_k(v, hlf);
        const unsigned short* p = &vt[(dt * 16 + nloc) * 64 + kk];
        bv0.d[v] = *(const unsigned int*)p;
        bv1.d[v] = *(const unsigned int*)(p + 32);
      }
      acc[dt] = wmma_bf16(p0.v, bv0.v, acc[dt]);
      acc[dt] = wmma_bf16(p1.v, bv1.v, acc[dt]);
    }
    __syncthreads();  // protect kt/vt before next chunk's async stores
  }

  // ---- normalize and store ctx (bf16, row-major [token][h*64+d]) ----
#pragma unroll
  for (int v = 0; v < 8; ++v) {
    float inv = 1.0f / rl[v];
    int m = v + 8 * hlf;
    size_t row = (size_t)(b * Sq + q0 + m);
#pragma unroll
    for (int dt = 0; dt < 4; ++dt)
      ctxb[row * HID + h * DH + dt * 16 + nloc] = f2bf(acc[dt][v] * inv);
  }
}

// ---------------------------------------------------------------------------
// Output projection: (4096 x 1024) x (1024 x 1024) -> f32 out.
// Block = 4 waves, each wave computes a 32x64 tile.
// ---------------------------------------------------------------------------
__global__ __launch_bounds__(128) void proj_gemm_kernel(
    const unsigned short* __restrict__ ctxb,  // 4096 x 1024 bf16
    const unsigned short* __restrict__ wptb,  // 1024 x 1024 bf16 (W^T)
    float* __restrict__ out) {                // 4096 x 1024 f32
  const int lane = threadIdx.x & 31;
  const int wave = threadIdx.x >> 5;
  const int nloc = lane & 15;
  const int hlf  = lane >> 4;
  const int m0   = blockIdx.y * 128 + wave * 32;
  const int c0   = blockIdx.x * 64;

  v8f acc[2][4] = {{{}, {}, {}, {}}, {{}, {}, {}, {}}};

  for (int kb = 0; kb < HID; kb += 32) {
    FragBF a[2];
#pragma unroll
    for (int r = 0; r < 2; ++r)
#pragma unroll
      for (int v = 0; v < 8; ++v) {
        int kk = frag_k(v, hlf);
        a[r].d[v] = *(const unsigned int*)(
            ctxb + (size_t)(m0 + r * 16 + nloc) * HID + kb + kk);
      }
#pragma unroll
    for (int t = 0; t < 4; ++t) {
      FragBF bw;  // B[k][n] = WT[n][k], contiguous in k
#pragma unroll
      for (int v = 0; v < 8; ++v) {
        int kk = frag_k(v, hlf);
        bw.d[v] = *(const unsigned int*)(
            wptb + (size_t)(c0 + t * 16 + nloc) * HID + kb + kk);
      }
      acc[0][t] = wmma_bf16(a[0].v, bw.v, acc[0][t]);
      acc[1][t] = wmma_bf16(a[1].v, bw.v, acc[1][t]);
    }
  }

#pragma unroll
  for (int t = 0; t < 4; ++t)
#pragma unroll
    for (int r = 0; r < 2; ++r)
#pragma unroll
      for (int v = 0; v < 8; ++v) {
        int m = r * 16 + v + 8 * hlf;
        out[(size_t)(m0 + m) * HID + c0 + t * 16 + nloc] = acc[r][t][v];
      }
}

}  // namespace

extern "C" void kernel_launch(void* const* d_in, const int* in_sizes, int n_in,
                              void* d_out, int out_size, void* d_ws,
                              size_t ws_size, hipStream_t stream) {
  (void)in_sizes; (void)n_in; (void)out_size; (void)ws_size;

  const float* x     = (const float*)d_in[0];  // (2, 2048, 1024)
  const float* sinu  = (const float*)d_in[1];  // (2, 2048, 32)
  const float* abias = (const float*)d_in[2];  // (2, 1, 2048, 2048)
  const float* wqkv  = (const float*)d_in[3];  // (1024, 48, 64)
  const float* qbias = (const float*)d_in[4];  // (48, 64)
  const float* wproj = (const float*)d_in[5];  // (16, 64, 1024)
  float* out = (float*)d_out;

  char* ws = (char*)d_ws;
  unsigned short* xb   = (unsigned short*)(ws + 0);                 //  8 MB
  unsigned short* wqtb = (unsigned short*)(ws + (8u  << 20));       //  6 MB
  unsigned short* wptb = (unsigned short*)(ws + (14u << 20));       //  2 MB
  unsigned short* Qb   = (unsigned short*)(ws + (16u << 20));       //  8 MB
  unsigned short* Kb   = (unsigned short*)(ws + (24u << 20));       //  8 MB
  unsigned short* VTb  = (unsigned short*)(ws + (32u << 20));       //  8 MB
  unsigned short* ctxb = (unsigned short*)(ws + (40u << 20));       //  8 MB

  cvt_kernel<<<2048, 256, 0, stream>>>(x, wqkv, wproj, xb, wqtb, wptb);

  qkv_gemm_kernel<<<dim3(NQKV / 64, MROWS / 128), 128, 0, stream>>>(
      xb, wqtb, qbias, sinu, Qb, Kb, VTb);

  attn_kernel<<<dim3(Sq / 64, NH, Bsz), 128, 0, stream>>>(
      Qb, Kb, VTb, abias, ctxb);

  proj_gemm_kernel<<<dim3(HID / 64, MROWS / 128), 128, 0, stream>>>(
      ctxb, wptb, out);
}
